// MultiHeadedAttention_11562051961106
// MI455X (gfx1250) — matrix operand order repaired and checked by reading, not compiled, not run on hardware
//
#include <hip/hip_runtime.h>
#include <hip/hip_bf16.h>

// Problem constants (from reference): B=2, S=4096, D=512, H=8, DK=64
#define B_  2
#define S_  4096
#define D_  512
#define H_  8
#define DK_ 64
#define M_  (B_ * S_)   // 8192 rows

typedef __attribute__((ext_vector_type(16))) __bf16 vbf16;
typedef __attribute__((ext_vector_type(8)))  float  v8f;

union Frag {
  vbf16 v;
  unsigned int u[8];
};

// round-to-nearest-even f32 -> bf16 (result in bits [31:16])
__device__ __forceinline__ unsigned int rne_hi(float f) {
  unsigned int x = __float_as_uint(f);
  return x + (0x7fffu + ((x >> 16) & 1u));
}
// pack two bf16 (lo, hi) into one dword with a single v_perm_b32
__device__ __forceinline__ unsigned int pack2bf(float lo, float hi) {
  return __builtin_amdgcn_perm(rne_hi(hi), rne_hi(lo), 0x07060302u);
}
__device__ __forceinline__ unsigned short f2bf(float f) {
  return (unsigned short)(rne_hi(f) >> 16);
}
__device__ __forceinline__ v8f wmma_bf16(const Frag& a, const Frag& b, v8f c) {
  return __builtin_amdgcn_wmma_f32_16x16x32_bf16(
      /*neg_a=*/false, a.v, /*neg_b=*/false, b.v,
      /*c_mod=*/(short)0, c, /*reuse_a=*/false, /*reuse_b=*/false);
}
__device__ __forceinline__ void wait_dscnt0() {
  asm volatile("s_wait_dscnt 0x0" ::: "memory");
}
__device__ __forceinline__ void wait_asynccnt0() {
  asm volatile("s_wait_asynccnt 0x0" ::: "memory");
}
// async copy 16B global -> LDS (per-lane), tracked by ASYNCcnt
__device__ __forceinline__ void async_g2l_b128(unsigned int lds_addr, const void* gaddr) {
  asm volatile("global_load_async_to_lds_b128 %0, %1, off"
               :: "v"(lds_addr), "v"(gaddr) : "memory");
}
__device__ __forceinline__ float rmax16(float v) {
  #pragma unroll
  for (int m = 1; m < 16; m <<= 1) v = fmaxf(v, __shfl_xor(v, m, 32));
  return v;
}
__device__ __forceinline__ float rsum16(float v) {
  #pragma unroll
  for (int m = 1; m < 16; m <<= 1) v += __shfl_xor(v, m, 32);
  return v;
}

// ---------------------------------------------------------------------------
// fp32 -> bf16 bulk conversion (memory bound; 8 elements / thread)
// ---------------------------------------------------------------------------
__global__ __launch_bounds__(256) void cvt_bf16_kernel(
    const float* __restrict__ src, unsigned short* __restrict__ dst, int n8)
{
  const int i = blockIdx.x * 256 + threadIdx.x;
  if (i >= n8) return;
  const float4* s4 = (const float4*)src;
  const float4 a = s4[2 * i];
  const float4 b = s4[2 * i + 1];
  uint4 o;
  o.x = pack2bf(a.x, a.y);
  o.y = pack2bf(a.z, a.w);
  o.z = pack2bf(b.x, b.y);
  o.w = pack2bf(b.z, b.w);
  ((uint4*)dst)[i] = o;
}

// ---------------------------------------------------------------------------
// GEMM: out[b,h,s,dk] = sum_k X[m,k] * W[n,k] + bias[n], n = h*64+dk, m=b*S+s
// X, W already bf16. One wave computes a 16(M) x 64(N) strip. All fragment
// loads for one K-chunk are batched before the 4 WMMAs.
// ---------------------------------------------------------------------------
__global__ __launch_bounds__(256) void proj_qkv_kernel(
    const unsigned short* __restrict__ Xbf,
    const unsigned short* __restrict__ Wbf,
    const float* __restrict__ bias, unsigned short* __restrict__ out)
{
  const int lane = threadIdx.x & 31;
  const int wid  = threadIdx.x >> 5;
  const int gw   = blockIdx.x * 8 + wid;      // 4096 waves total
  const int mtile  = gw >> 3;                 // 0..511
  const int nstrip = gw & 7;                  // head index (64 cols == DK)
  const int half = lane >> 4;
  const int lrow = lane & 15;
  const int kbA = half ? 8 : 0;               // A-frag K base (u32 units)
  const int kbB = half ? 8 : 0;               // B-frag K base (u32 units)

  v8f acc[4] = {};
  const unsigned int* xrow =
      (const unsigned int*)(Xbf + (size_t)(mtile * 16 + lrow) * D_);
  const unsigned int* wrow[4];
  #pragma unroll
  for (int sub = 0; sub < 4; ++sub)
    wrow[sub] = (const unsigned int*)(Wbf + (size_t)(nstrip * 64 + sub * 16 + lrow) * D_);

  for (int k0 = 0; k0 < D_; k0 += 32) {
    const int kk = k0 >> 1;                   // u32 offset of this K chunk
    __builtin_prefetch(xrow + kk + 32, 0, 3);
    Frag a;
    Frag bfr[4];
    #pragma unroll
    for (int j = 0; j < 4; ++j) {
      a.u[j]     = xrow[kk + kbA + j];
      a.u[4 + j] = xrow[kk + kbA + 4 + j];
    }
    #pragma unroll
    for (int sub = 0; sub < 4; ++sub)
      #pragma unroll
      for (int j = 0; j < 8; ++j) bfr[sub].u[j] = wrow[sub][kk + kbB + j];
    #pragma unroll
    for (int sub = 0; sub < 4; ++sub)
      acc[sub] = wmma_bf16(a, bfr[sub], acc[sub]);
  }

  const int h = nstrip;
  #pragma unroll
  for (int sub = 0; sub < 4; ++sub) {
    const int dk = sub * 16 + lrow;
    const float bn = bias[h * 64 + dk];
    #pragma unroll
    for (int r = 0; r < 8; ++r) {
      const int m = mtile * 16 + r + half * 8;
      const int bb = m >> 12;           // m / S_
      const int s  = m & (S_ - 1);
      out[(((size_t)(bb * H_ + h) * S_ + s) * DK_) + dk] = f2bf(acc[sub][r] + bn);
    }
  }
}

// ---------------------------------------------------------------------------
// Output projection: attn(bf16,[M_,D_]) x Wo^T(bf16) + bo -> fp32 out [M_,D_]
// ---------------------------------------------------------------------------
__global__ __launch_bounds__(256) void proj_out_kernel(
    const unsigned short* __restrict__ Xbf,
    const unsigned short* __restrict__ Wbf,
    const float* __restrict__ bias, float* __restrict__ out)
{
  const int lane = threadIdx.x & 31;
  const int wid  = threadIdx.x >> 5;
  const int gw   = blockIdx.x * 8 + wid;
  const int mtile  = gw >> 3;
  const int nstrip = gw & 7;
  const int half = lane >> 4;
  const int lrow = lane & 15;
  const int kbA = half ? 8 : 0;
  const int kbB = half ? 8 : 0;

  v8f acc[4] = {};
  const unsigned int* xrow =
      (const unsigned int*)(Xbf + (size_t)(mtile * 16 + lrow) * D_);
  const unsigned int* wrow[4];
  #pragma unroll
  for (int sub = 0; sub < 4; ++sub)
    wrow[sub] = (const unsigned int*)(Wbf + (size_t)(nstrip * 64 + sub * 16 + lrow) * D_);

  for (int k0 = 0; k0 < D_; k0 += 32) {
    const int kk = k0 >> 1;
    __builtin_prefetch(xrow + kk + 32, 0, 3);
    Frag a;
    Frag bfr[4];
    #pragma unroll
    for (int j = 0; j < 4; ++j) {
      a.u[j]     = xrow[kk + kbA + j];
      a.u[4 + j] = xrow[kk + kbA + 4 + j];
    }
    #pragma unroll
    for (int sub = 0; sub < 4; ++sub)
      #pragma unroll
      for (int j = 0; j < 8; ++j) bfr[sub].u[j] = wrow[sub][kk + kbB + j];
    #pragma unroll
    for (int sub = 0; sub < 4; ++sub)
      acc[sub] = wmma_bf16(a, bfr[sub], acc[sub]);
  }

  #pragma unroll
  for (int sub = 0; sub < 4; ++sub) {
    const int n = nstrip * 64 + sub * 16 + lrow;
    const float bn = bias[n];
    #pragma unroll
    for (int r = 0; r < 8; ++r) {
      const int m = mtile * 16 + r + half * 8;
      out[(size_t)m * D_ + n] = acc[sub][r] + bn;
    }
  }
}

// ---------------------------------------------------------------------------
// Causal flash attention. 1 wave = 16 query rows; key tiles of 32.
// V tiles are copied global->LDS with async b128 ops (ASYNCcnt), overlapped
// with the Q*K^T WMMAs.
// ---------------------------------------------------------------------------
__global__ __launch_bounds__(128) void attn_kernel(
    const unsigned short* __restrict__ Q,
    const unsigned short* __restrict__ K,
    const unsigned short* __restrict__ V,
    unsigned short* __restrict__ O)
{
  // per-wave: Vr row-major [32 keys][64 feats] u16 (1024 u32)
  //           P  [16][34] u16 (272 u32)            -> 1296 u32 per wave
  __shared__ unsigned int smem[4 * 1296];

  const int lane = threadIdx.x & 31;
  const int wid  = threadIdx.x >> 5;
  const int half = lane >> 4;
  const int lrow = lane & 15;

  const int bid = blockIdx.x;           // B_*H_*(S_/64) = 1024
  const int qb = bid & 63;
  const int h  = (bid >> 6) & 7;
  const int b  = bid >> 9;
  const int qbase = qb * 64 + wid * 16;

  const unsigned short* Qh = Q + (size_t)(b * H_ + h) * S_ * DK_;
  const unsigned short* Kh = K + (size_t)(b * H_ + h) * S_ * DK_;
  const unsigned short* Vh = V + (size_t)(b * H_ + h) * S_ * DK_;

  unsigned int* sw = smem + wid * 1296;
  const unsigned short* Vr = (const unsigned short*)sw;      // [32][64] u16
  unsigned short* Pb = (unsigned short*)(sw + 1024);         // [16][34] u16
  const unsigned int* Pb32 = sw + 1024;
  const unsigned int vlds = (unsigned int)(unsigned long long)(const void*)sw;

  // Q fragments (fixed for whole loop), A layout, K split at feat 0/32
  Frag qa[2];
  {
    const unsigned int* qp = (const unsigned int*)(Qh + (size_t)(qbase + lrow) * DK_);
    const int kb = half ? 8 : 0;
    #pragma unroll
    for (int kc = 0; kc < 2; ++kc)
      #pragma unroll
      for (int j = 0; j < 4; ++j) {
        qa[kc].u[j]     = qp[kc * 16 + kb + j];
        qa[kc].u[4 + j] = qp[kc * 16 + kb + 4 + j];
      }
  }

  v8f o[4] = {};
  float rowmax[8], rowsum[8];
  #pragma unroll
  for (int r = 0; r < 8; ++r) { rowmax[r] = -1e30f; rowsum[r] = 0.f; }

  const int qend = qbase + 15;
  for (int kt = 0; kt <= qend; kt += 32) {
    // WAR fence: previous iteration's LDS reads must retire before we
    // overwrite Vr (async engine) and Pb (ds stores).
    wait_dscnt0();

    // async copy V tile (32 keys x 64 feats = 4KB contiguous) into LDS,
    // overlapped with the score WMMAs below.
    {
      const char* gsrc = (const char*)(Vh + (size_t)kt * DK_);
      #pragma unroll
      for (int it = 0; it < 8; ++it)
        async_g2l_b128(vlds + (unsigned)(lane * 16 + it * 512),
                       gsrc + lane * 16 + it * 512);
    }

    // scores: S = Q * K^T  (two 16x16 tiles covering 32 keys)
    v8f s[2];
    #pragma unroll
    for (int g = 0; g < 2; ++g) {
      v8f accs = {};
      const int key = kt + g * 16 + lrow;
      const unsigned int* kp = (const unsigned int*)(Kh + (size_t)key * DK_);
      const int kb = half ? 8 : 0;      // u32
      #pragma unroll
      for (int kc = 0; kc < 2; ++kc) {
        Frag kf;
        #pragma unroll
        for (int j = 0; j < 8; ++j) kf.u[j] = kp[kc * 16 + kb + j];
        accs = wmma_bf16(qa[kc], kf, accs);
      }
      s[g] = accs;
    }

    // scale + causal mask
    #pragma unroll
    for (int g = 0; g < 2; ++g) {
      const int key = kt + g * 16 + lrow;
      #pragma unroll
      for (int r = 0; r < 8; ++r) {
        const int q = qbase + r + half * 8;
        const float vv = s[g][r] * 0.125f;   // 1/sqrt(64)
        s[g][r] = (key <= q) ? vv : -1e30f;
      }
    }

    // online softmax update
    float scl[8];
    #pragma unroll
    for (int r = 0; r < 8; ++r) {
      const float tmax = rmax16(fmaxf(s[0][r], s[1][r]));
      const float mnew = fmaxf(rowmax[r], tmax);
      const float sc0  = __expf(rowmax[r] - mnew);
      const float p0   = __expf(s[0][r] - mnew);
      const float p1   = __expf(s[1][r] - mnew);
      s[0][r] = p0; s[1][r] = p1;
      rowsum[r] = rowsum[r] * sc0 + rsum16(p0 + p1);
      rowmax[r] = mnew;
      scl[r] = sc0;
    }
    #pragma unroll
    for (int n = 0; n < 4; ++n)
      #pragma unroll
      for (int r = 0; r < 8; ++r) o[n][r] *= scl[r];

    // P (C layout) -> LDS bf16
    #pragma unroll
    for (int g = 0; g < 2; ++g)
      #pragma unroll
      for (int r = 0; r < 8; ++r)
        Pb[(r + half * 8) * 34 + g * 16 + lrow] = f2bf(s[g][r]);
    wait_dscnt0();     // P visible to all lanes
    wait_asynccnt0();  // V tile resident in LDS

    // P as A fragment (16x32)
    Frag pa;
    {
      const int kb = half ? 8 : 0;
      const int rb = lrow * 17;
      #pragma unroll
      for (int j = 0; j < 4; ++j) {
        pa.u[j]     = Pb32[rb + kb + j];
        pa.u[4 + j] = Pb32[rb + kb + 4 + j];
      }
    }
    // V columns as B fragments (gather u16 pairs from row-major Vr)
    {
      const int kb = half ? 16 : 0;     // key base for this lane-half
      #pragma unroll
      for (int n = 0; n < 4; ++n) {
        const int f = n * 16 + lrow;    // feature column
        Frag vf;
        #pragma unroll
        for (int j = 0; j < 8; ++j) {
          const unsigned int lo = Vr[(kb + 2 * j) * 64 + f];
          const unsigned int hi = Vr[(kb + 2 * j + 1) * 64 + f];
          vf.u[j] = lo | (hi << 16);
        }
        o[n] = wmma_bf16(pa, vf, o[n]);
      }
    }
  }

  // normalize and store (bf16, [B,S,D] with head-concat columns)
  float inv[8];
  #pragma unroll
  for (int r = 0; r < 8; ++r) inv[r] = 1.0f / rowsum[r];
  #pragma unroll
  for (int n = 0; n < 4; ++n) {
    const int col = h * 64 + n * 16 + lrow;
    #pragma unroll
    for (int r = 0; r < 8; ++r) {
      const int m = b * S_ + qbase + r + half * 8;
      O[(size_t)m * D_ + col] = f2bf(o[n][r] * inv[r]);
    }
  }
}

// ---------------------------------------------------------------------------
extern "C" void kernel_launch(void* const* d_in, const int* in_sizes, int n_in,
                              void* d_out, int out_size, void* d_ws, size_t ws_size,
                              hipStream_t stream) {
  const float* query = (const float*)d_in[0];
  const float* key   = (const float*)d_in[1];
  const float* value = (const float*)d_in[2];
  const float* Wq = (const float*)d_in[3];
  const float* bq = (const float*)d_in[4];
  const float* Wk = (const float*)d_in[5];
  const float* bk = (const float*)d_in[6];
  const float* Wv = (const float*)d_in[7];
  const float* bv = (const float*)d_in[8];
  const float* Wo = (const float*)d_in[9];
  const float* bo = (const float*)d_in[10];
  // d_in[11] is the causal mask; causality is applied analytically.

  const size_t xelems = (size_t)M_ * D_;        // 4M activations
  const size_t welems = (size_t)D_ * D_;        // 256K weights
  unsigned short* xbf  = (unsigned short*)d_ws;       // reused per projection
  unsigned short* wbf  = xbf + xelems;                // reused per projection
  unsigned short* qbuf = wbf + welems;
  unsigned short* kbuf = qbuf + xelems;
  unsigned short* vbuf = kbuf + xelems;
  unsigned short* abuf = vbuf + xelems;
  float* out = (float*)d_out;

  const int xn8 = (int)(xelems / 8);
  const int wn8 = (int)(welems / 8);
  const dim3 cvtBx((xn8 + 255) / 256), cvtBw((wn8 + 255) / 256), cvtT(256);
  const dim3 gProj(512), bProj(256);

  // Q projection
  cvt_bf16_kernel<<<cvtBx, cvtT, 0, stream>>>(query, xbf, xn8);
  cvt_bf16_kernel<<<cvtBw, cvtT, 0, stream>>>(Wq, wbf, wn8);
  proj_qkv_kernel<<<gProj, bProj, 0, stream>>>(xbf, wbf, bq, qbuf);
  // K projection
  cvt_bf16_kernel<<<cvtBx, cvtT, 0, stream>>>(key, xbf, xn8);
  cvt_bf16_kernel<<<cvtBw, cvtT, 0, stream>>>(Wk, wbf, wn8);
  proj_qkv_kernel<<<gProj, bProj, 0, stream>>>(xbf, wbf, bk, kbuf);
  // V projection
  cvt_bf16_kernel<<<cvtBx, cvtT, 0, stream>>>(value, xbf, xn8);
  cvt_bf16_kernel<<<cvtBw, cvtT, 0, stream>>>(Wv, wbf, wn8);
  proj_qkv_kernel<<<gProj, bProj, 0, stream>>>(xbf, wbf, bv, vbuf);

  // causal attention
  attn_kernel<<<dim3(B_ * H_ * (S_ / 64)), dim3(128), 0, stream>>>(qbuf, kbuf, vbuf, abuf);

  // output projection (fp32 result)
  cvt_bf16_kernel<<<cvtBw, cvtT, 0, stream>>>(Wo, wbf, wn8);
  proj_out_kernel<<<gProj, bProj, 0, stream>>>(abuf, wbf, bo, out);
}
